// GraphNeuralNetwork_41085657153662
// MI455X (gfx1250) — matrix operand order repaired and checked by reading, not compiled, not run on hardware
//
#include <hip/hip_runtime.h>

typedef __attribute__((ext_vector_type(16))) _Float16 v16h;
typedef __attribute__((ext_vector_type(8)))  _Float16 v8h;
typedef __attribute__((ext_vector_type(8)))  float    v8f;
typedef __attribute__((ext_vector_type(4)))  float    v4f;
typedef __attribute__((address_space(3)))    _Float16 lds_f16;

constexpr int NN = 50000;   // nodes
constexpr int EE = 400000;  // edges
constexpr int GG = 256;     // graphs

// ---------------- degree / norm ----------------
__global__ void k_init_deg(float* deg, int n) {
    int i = blockIdx.x * blockDim.x + threadIdx.x;
    if (i < n) deg[i] = 1.0f;                      // self-loop
}

__global__ void k_deg_accum(float* deg, const int* __restrict__ dst, int e) {
    int i = blockIdx.x * blockDim.x + threadIdx.x;
    if (i < e) atomicAdd(&deg[dst[i]], 1.0f);
}

__global__ void k_dis(float* dis, const float* __restrict__ deg, int n) {
    int i = blockIdx.x * blockDim.x + threadIdx.x;
    if (i < n) dis[i] = rsqrtf(deg[i]);
}

// ---------------- activations: f32 -> padded f16 (optional fused relu) ----------------
__global__ void k_act_convert(_Float16* __restrict__ Xh, const float* __restrict__ src,
                              int F, int Kp, int n, int relu) {
    int i = blockIdx.x * blockDim.x + threadIdx.x;
    if (i < n * Kp) {
        int node = i / Kp, k = i % Kp;
        float v = 0.0f;
        if (k < F) {
            v = src[(size_t)node * F + k];
            if (relu) v = fmaxf(v, 0.0f);
        }
        Xh[i] = (_Float16)v;
    }
}

// ---------------- weights: f32 -> tile-major f16 (exact WMMA B-fragment order) ------
// Whp[((ct*nkt + kt)*512) + n*32 + k] = W[(kt*32+k), (ct*16+n)]
__global__ void k_pack_w(_Float16* __restrict__ Whp, const float* __restrict__ W,
                         int Kin, int Nout, int Kp, int Np) {
    int i = blockIdx.x * blockDim.x + threadIdx.x;
    if (i < Kp * Np) {
        int tile = i >> 9, r = i & 511;
        int n = r >> 5, k = r & 31;
        int nkt = Kp >> 5;
        int ct = tile / nkt, kt = tile - ct * nkt;
        int col = ct * 16 + n, kk = kt * 32 + k;
        Whp[i] = (kk < Kin && col < Nout) ? (_Float16)W[(size_t)kk * Nout + col]
                                          : (_Float16)0.0f;
    }
}

// ---------------- WMMA GEMM: Out[N,Nout] = Xh[N,Kp](f16) @ W ----------------
// 8 waves/block; wave w -> rows [blk.x*128 + w*16, +16), cols [blk.y*16, +16).
// B tiles streamed through double-buffered LDS with async global->LDS copies
// prefetched one K-step ahead (overlaps the async engine with v_wmma).
__global__ __launch_bounds__(256) void k_gemm_wmma(
    const _Float16* __restrict__ Xh, const _Float16* __restrict__ Whp,
    float* __restrict__ Out, int Kp, int Nout)
{
    __shared__ __align__(64) _Float16 Bs[2][512];  // [buf][n:16][k:32]
    const int tid  = threadIdx.x;
    const int wave = tid >> 5;
    const int lane = tid & 31;
    const int hh   = lane >> 4;                    // lane half (0/1)
    const int l16  = lane & 15;
    const int rowb = blockIdx.x * 128 + wave * 16;
    int arow = rowb + l16;
    if (arow >= NN) arow = NN - 1;                 // clamp loads; stores are guarded
    const _Float16* __restrict__ Xrow = Xh + (size_t)arow * Kp;
    const int nkt = Kp >> 5;
    const _Float16* __restrict__ gW = Whp + (size_t)blockIdx.y * nkt * 512;

    // prologue: async-stage tile 0 into buffer 0 (waves 0-1: 64 x b128 chunks)
    if (tid < 64) {
        const _Float16* gsrc = gW + tid * 8;
        unsigned loff = (unsigned)(uintptr_t)(lds_f16*)(&Bs[0][tid * 8]);
        asm volatile("global_load_async_to_lds_b128 %0, %1, off"
                     :: "v"(loff), "v"(gsrc) : "memory");
    }

    v8f c = {};
    for (int kt = 0; kt < nkt; ++kt) {
        // prefetch tile kt+1 (last iter: redundant re-issue keeps wait imm constant)
        if (tid < 64) {
            int nxt = (kt + 1 < nkt) ? kt + 1 : nkt - 1;
            const _Float16* gsrc = gW + (size_t)nxt * 512 + tid * 8;
            unsigned loff = (unsigned)(uintptr_t)(lds_f16*)(&Bs[(kt + 1) & 1][tid * 8]);
            asm volatile("global_load_async_to_lds_b128 %0, %1, off"
                         :: "v"(loff), "v"(gsrc) : "memory");
            // async loads complete in order: <=1 outstanding => tile kt has landed
            asm volatile("s_wait_asynccnt 1" ::: "memory");
        }
        __syncthreads();

        // ---- A fragment: two 16B chunks, k = 16*half + s (matches B contiguous order) ----
        const int k0 = kt * 32;
        v8h a0 = *reinterpret_cast<const v8h*>(Xrow + k0 + hh * 16);
        v8h a1 = *reinterpret_cast<const v8h*>(Xrow + k0 + hh * 16 + 8);
        v16h a;
#pragma unroll
        for (int j = 0; j < 8; ++j) { a[j] = a0[j]; a[8 + j] = a1[j]; }

        // ---- B fragment: lane = column, lane-half = K half; 32B from LDS ----
        v16h b = *reinterpret_cast<const v16h*>(&Bs[kt & 1][l16 * 32 + hh * 16]);

        c = __builtin_amdgcn_wmma_f32_16x16x32_f16(false, a, false, b, (short)0, c,
                                                   false, false);
        __syncthreads();   // WAR: all waves done with Bs[kt&1] before it is re-filled
    }
    // D layout: VGPR r -> M = r + half*8, N = lane%16
#pragma unroll
    for (int r = 0; r < 8; ++r) {
        int row = rowb + r + hh * 8;
        int col = blockIdx.y * 16 + l16;
        if (row < NN && col < Nout) Out[(size_t)row * Nout + col] = c[r];
    }
}

// ---------------- aggregation: out = dis^2 * h + bias, then edge scatter ----------------
__global__ void k_self_bias(float* __restrict__ out, const float* __restrict__ h,
                            const float* __restrict__ dis, const float* __restrict__ bias,
                            int F, int n) {
    int i = blockIdx.x * blockDim.x + threadIdx.x;
    if (i < n * F) {
        int node = i / F, f = i % F;
        float d = dis[node];
        out[i] = d * d * h[i] + bias[f];
    }
}

__global__ void k_scatter(float* __restrict__ out, const float* __restrict__ h,
                          const float* __restrict__ dis, const int* __restrict__ src,
                          const int* __restrict__ dst, int F4, int F, int e) {
    int i = blockIdx.x * blockDim.x + threadIdx.x;
    int total = e * F4;
    if (i < total) {
        int ed = i / F4, cidx = i - ed * F4;
        int s = src[ed], d = dst[ed];
        float norm = dis[s] * dis[d];
        v4f hv = *reinterpret_cast<const v4f*>(&h[(size_t)s * F + 4 * cidx]);
        float* o = &out[(size_t)d * F + 4 * cidx];
        atomicAdd(o + 0, norm * hv.x);
        atomicAdd(o + 1, norm * hv.y);
        atomicAdd(o + 2, norm * hv.z);
        atomicAdd(o + 3, norm * hv.w);
    }
}

// ---------------- pooling (relu fused) + head ----------------
__global__ void k_zero(float* p, int total) {
    int i = blockIdx.x * blockDim.x + threadIdx.x;
    if (i < total) p[i] = 0.0f;
}

__global__ void k_cnt(float* cnt, const int* __restrict__ batch, int n) {
    int i = blockIdx.x * blockDim.x + threadIdx.x;
    if (i < n) atomicAdd(&cnt[batch[i]], 1.0f);
}

__global__ void k_pool(float* __restrict__ sums, const float* __restrict__ h,
                       const int* __restrict__ batch, int n) {
    int i = blockIdx.x * blockDim.x + threadIdx.x;
    if (i < n * 25) {
        int node = i / 25, cidx = i - node * 25;
        int g = batch[node];
        v4f hv = *reinterpret_cast<const v4f*>(&h[(size_t)node * 100 + 4 * cidx]);
        float* o = &sums[(size_t)g * 100 + 4 * cidx];
        atomicAdd(o + 0, fmaxf(hv.x, 0.0f));
        atomicAdd(o + 1, fmaxf(hv.y, 0.0f));
        atomicAdd(o + 2, fmaxf(hv.z, 0.0f));
        atomicAdd(o + 3, fmaxf(hv.w, 0.0f));
    }
}

__global__ void k_head1(float* __restrict__ tmp, const float* __restrict__ sums,
                        const float* __restrict__ cnt, const float* __restrict__ Wo,
                        const float* __restrict__ bo) {
    int i = blockIdx.x * blockDim.x + threadIdx.x;
    if (i < GG * 50) {
        int g = i / 50, j = i - (i / 50) * 50;
        float inv = 1.0f / fmaxf(cnt[g], 1.0f);
        float acc = bo[j];
        for (int k = 0; k < 100; ++k) acc += sums[g * 100 + k] * inv * Wo[k * 50 + j];
        tmp[i] = fmaxf(acc, 0.0f);
    }
}

__global__ void k_head2(float* __restrict__ out, const float* __restrict__ tmp,
                        const float* __restrict__ Wo2, const float* __restrict__ bo2) {
    int g = blockIdx.x * blockDim.x + threadIdx.x;
    if (g < GG) {
        float acc = bo2[0];
        for (int j = 0; j < 50; ++j) acc += tmp[g * 50 + j] * Wo2[j];
        out[g] = acc;
    }
}

// ---------------- host-side layer driver ----------------
static void run_layer(const _Float16* Xh, int Kp, const float* W, const float* bias,
                      int Kin, int Nout, float* hA, float* hB, _Float16* Whp,
                      const float* dis, const int* src, const int* dst,
                      hipStream_t stream)
{
    int Np = (Nout + 15) & ~15;
    int cw = Kp * Np;
    k_pack_w<<<(cw + 255) / 256, 256, 0, stream>>>(Whp, W, Kin, Nout, Kp, Np);

    dim3 gg((NN + 127) / 128, Np / 16);
    k_gemm_wmma<<<gg, 256, 0, stream>>>(Xh, Whp, hA, Kp, Nout);

    int totSB = NN * Nout;
    k_self_bias<<<(totSB + 255) / 256, 256, 0, stream>>>(hB, hA, dis, bias, Nout, NN);

    int F4 = Nout / 4;
    long totS = (long)EE * F4;
    k_scatter<<<(unsigned)((totS + 255) / 256), 256, 0, stream>>>(hB, hA, dis, src, dst,
                                                                  F4, Nout, EE);
}

extern "C" void kernel_launch(void* const* d_in, const int* in_sizes, int n_in,
                              void* d_out, int out_size, void* d_ws, size_t ws_size,
                              hipStream_t stream) {
    const float* x     = (const float*)d_in[0];       // [N,42]
    const int*   ei    = (const int*)d_in[1];         // [2,E]
    const int*   batch = (const int*)d_in[2];         // [N]
    const float* W1 = (const float*)d_in[3];  const float* b1 = (const float*)d_in[4];
    const float* W2 = (const float*)d_in[5];  const float* b2 = (const float*)d_in[6];
    const float* W3 = (const float*)d_in[7];  const float* b3 = (const float*)d_in[8];
    const float* W4 = (const float*)d_in[9];  const float* b4 = (const float*)d_in[10];
    const float* Wo = (const float*)d_in[11]; const float* bo = (const float*)d_in[12];
    const float* Wo2 = (const float*)d_in[13]; const float* bo2 = (const float*)d_in[14];
    const int* src = ei;
    const int* dst = ei + EE;
    float* out = (float*)d_out;                        // [G,1]

    // workspace layout (floats first; f16 region 16B-aligned by construction)
    float* ws   = (float*)d_ws;
    float* hA   = ws;                                  // N*400 (GEMM output)
    float* hB   = hA + (size_t)NN * 400;               // N*400 (aggregated, pre-relu)
    float* deg  = hB + (size_t)NN * 400;               // N
    float* dis  = deg + NN;                            // N
    float* sums = dis + NN;                            // G*100
    float* cnt  = sums + (size_t)GG * 100;             // G
    float* tmp  = cnt + GG;                            // G*50
    _Float16* Xh  = (_Float16*)(tmp + (size_t)GG * 50); // N*416 max (padded f16 acts)
    _Float16* Whp = Xh + (size_t)NN * 416;              // <= 416*416 packed f16 weights

    // degree / normalization
    k_init_deg<<<(NN + 255) / 256, 256, 0, stream>>>(deg, NN);
    k_deg_accum<<<(EE + 255) / 256, 256, 0, stream>>>(deg, dst, EE);
    k_dis<<<(NN + 255) / 256, 256, 0, stream>>>(dis, deg, NN);

    // layer 1: x (K=42 -> Kp=64)
    k_act_convert<<<(NN * 64 + 255) / 256, 256, 0, stream>>>(Xh, x, 42, 64, NN, 0);
    run_layer(Xh, 64, W1, b1, 42, 100, hA, hB, Whp, dis, src, dst, stream);

    // layer 2: relu(hB) (K=100 -> Kp=128)
    k_act_convert<<<(NN * 128 + 255) / 256, 256, 0, stream>>>(Xh, hB, 100, 128, NN, 1);
    run_layer(Xh, 128, W2, b2, 100, 400, hA, hB, Whp, dis, src, dst, stream);

    // layer 3: relu(hB) (K=400 -> Kp=416)
    k_act_convert<<<(NN * 416 + 255) / 256, 256, 0, stream>>>(Xh, hB, 400, 416, NN, 1);
    run_layer(Xh, 416, W3, b3, 400, 200, hA, hB, Whp, dis, src, dst, stream);

    // layer 4: relu(hB) (K=200 -> Kp=224); hB ends as pre-relu h4
    k_act_convert<<<(NN * 224 + 255) / 256, 256, 0, stream>>>(Xh, hB, 200, 224, NN, 1);
    run_layer(Xh, 224, W4, b4, 200, 100, hA, hB, Whp, dis, src, dst, stream);

    // global mean pool over relu(h4)
    k_zero<<<(GG * 100 + GG + 255) / 256, 256, 0, stream>>>(sums, GG * 100 + GG);
    k_cnt<<<(NN + 255) / 256, 256, 0, stream>>>(cnt, batch, NN);
    k_pool<<<(NN * 25 + 255) / 256, 256, 0, stream>>>(sums, hB, batch, NN);

    // MLP head
    k_head1<<<(GG * 50 + 255) / 256, 256, 0, stream>>>(tmp, sums, cnt, Wo, bo);
    k_head2<<<(GG + 255) / 256, 256, 0, stream>>>(out, tmp, Wo2, bo2);
}
